// Transformer_fPEPS_Model_batchedAttn_36721970380871
// MI455X (gfx1250) — matrix operand order repaired and checked by reading, not compiled, not run on hardware
//
#include <hip/hip_runtime.h>
#include <math.h>

// ---------------- static problem constants ----------------
#define BATCH 2048
#define NS    36
#define EMB   128
#define HEADS 8
#define DH    16
#define HID   512
#define MAXS  512
#define TOTAL 10368
#define ETA   0.1f
#define MTOK  (BATCH * NS)        // 73728 rows for all GEMMs

typedef _Float16 v16h __attribute__((ext_vector_type(16)));
typedef float    v8f  __attribute__((ext_vector_type(8)));
typedef unsigned v4u  __attribute__((ext_vector_type(4)));

// per-site PEPS parameter offsets / sizes (with phys dim included)
__constant__ int c_offs[36] = {
    0,32,160,288,416,544,
    576,704,1216,1728,2240,2752,
    2880,3008,3520,4032,4544,5056,
    5184,5312,5824,6336,6848,7360,
    7488,7616,8128,8640,9152,9664,
    9792,9824,9952,10080,10208,10336};
__constant__ int c_sizes[36] = {
    32,128,128,128,128,32,
    128,512,512,512,512,128,
    128,512,512,512,512,128,
    128,512,512,512,512,128,
    128,512,512,512,512,128,
    32,128,128,128,128,32};

// ---------------- weight convert + transpose (f32 [K][N] -> f16 [N][K]) ----
__global__ void k_convT(const float* __restrict__ src, _Float16* __restrict__ dst,
                        int K, int N) {
    int idx = blockIdx.x * blockDim.x + threadIdx.x;
    if (idx >= K * N) return;
    int k = idx / N, n = idx % N;
    dst[(size_t)n * K + k] = (_Float16)src[idx];
}

// ---------------- embedding + positional ----------------
__global__ void k_embed(const int* __restrict__ x, const float* __restrict__ emb,
                        const float* __restrict__ pos,
                        float* __restrict__ hf32, _Float16* __restrict__ hf16) {
    size_t idx = (size_t)blockIdx.x * blockDim.x + threadIdx.x;
    if (idx >= (size_t)MTOK * EMB) return;
    int t = (int)(idx >> 7);          // token
    int e = (int)(idx & 127);
    float v = emb[x[t] * EMB + e] + pos[(t % NS) * EMB + e];
    hf32[idx] = v;
    hf16[idx] = (_Float16)v;
}

// ---------------- WMMA GEMM: C[M,N] = A[M,K] * Bt[N,K]^T + epilogue --------
// Each wave computes a 16x32 output tile (two 16x16 WMMA tiles sharing the
// A fragment). Fully unrolled K loop; compile-time epilogue via MODE:
//  MODE 0: +bias -> f32 out (qkv)
//  MODE 1: +bias +resid(f32) -> f16 out (attn output proj + residual)
//  MODE 2: +bias, exact GELU -> f16 out (MLP hidden)
//  MODE 3: +bias, per-site trim: vecs[b,off+n] = ftn[off+n] + ETA*val
template <int N, int K, int MODE>
__global__ __launch_bounds__(256)
void k_gemm(const _Float16* __restrict__ A, const _Float16* __restrict__ Bt,
            const float* __restrict__ bias, const float* __restrict__ resid,
            const float* __restrict__ ftn,
            float* __restrict__ outf, _Float16* __restrict__ outh) {
    constexpr int  TILES_N2 = N / 32;
    constexpr long TOTAL_TILES = (long)(MTOK / 16) * TILES_N2;

    int wave = threadIdx.x >> 5;
    int lane = threadIdx.x & 31;
    int ln = lane & 15;           // wave32: low nibble picks row/col in tile
    int g  = lane >> 4;           // lane group (0: lanes 0-15, 1: lanes 16-31)

    long tile = (long)blockIdx.x * 8 + wave;
    if (tile >= TOTAL_TILES) return;
    int m0 = (int)(tile / TILES_N2) << 4;
    int n0 = (int)(tile % TILES_N2) << 5;

    union FragU { v16h h; v4u u[2]; };

    const _Float16* arow  = A  + (size_t)(m0 + ln) * K;
    const _Float16* brow0 = Bt + (size_t)(n0 + ln) * K;
    const _Float16* brow1 = brow0 + (size_t)16 * K;

    __builtin_prefetch(arow, 0, 3);
    __builtin_prefetch(brow0, 0, 3);
    __builtin_prefetch(brow1, 0, 3);

    v8f acc[2];
    acc[0] = (v8f){0.f, 0.f, 0.f, 0.f, 0.f, 0.f, 0.f, 0.f};
    acc[1] = (v8f){0.f, 0.f, 0.f, 0.f, 0.f, 0.f, 0.f, 0.f};

#pragma unroll
    for (int k0 = 0; k0 < K; k0 += 32) {
        FragU a, b0, b1;
        // A frag (K contiguous per lane-half): K = {g*16 .. g*16+15}
        a.u[0]  = *(const v4u*)(arow + k0 + g * 16);
        a.u[1]  = *(const v4u*)(arow + k0 + g * 16 + 8);
        // B frag (K contiguous per lane-half): K = {g*16 .. g*16+15}
        b0.u[0] = *(const v4u*)(brow0 + k0 + g * 16);
        b0.u[1] = *(const v4u*)(brow0 + k0 + g * 16 + 8);
        b1.u[0] = *(const v4u*)(brow1 + k0 + g * 16);
        b1.u[1] = *(const v4u*)(brow1 + k0 + g * 16 + 8);
        acc[0] = __builtin_amdgcn_wmma_f32_16x16x32_f16(
                     false, a.h, false, b0.h, (short)0, acc[0], false, false);
        acc[1] = __builtin_amdgcn_wmma_f32_16x16x32_f16(
                     false, a.h, false, b1.h, (short)0, acc[1], false, false);
    }

#pragma unroll
    for (int half = 0; half < 2; ++half) {
        int n = n0 + half * 16 + ln;
        float bn = bias[n];
#pragma unroll
        for (int r = 0; r < 8; ++r) {
            int m = m0 + r + g * 8;   // C/D layout: VGPR r; lanes16-31 M=r+8
            float v = acc[half][r] + bn;
            if constexpr (MODE == 0) {
                outf[(size_t)m * N + n] = v;
            } else if constexpr (MODE == 1) {
                outh[(size_t)m * N + n] =
                    (_Float16)(resid[(size_t)m * N + n] + v);
            } else if constexpr (MODE == 2) {
                float ge = 0.5f * v * (1.0f + erff(v * 0.70710678118654752f));
                outh[(size_t)m * N + n] = (_Float16)ge;
            } else {
                int s = m % NS;
                if (n < c_sizes[s]) {
                    size_t b = (size_t)(m / NS);
                    int off = c_offs[s] + n;
                    outf[b * TOTAL + off] = ftn[off] + ETA * v;
                }
            }
        }
    }
}

// ---------------- attention: per (batch, head) ----------------
__global__ __launch_bounds__(64)
void k_attn(const float* __restrict__ qkv, _Float16* __restrict__ o16) {
    int bh = blockIdx.x;
    int b = bh >> 3, h = bh & 7;
    __shared__ float sq[NS][DH], sk[NS][DH], sv[NS][DH];
    int tid = threadIdx.x;
    for (int t = tid; t < NS * DH; t += 64) {
        int s = t >> 4, d = t & 15;
        size_t base = ((size_t)b * NS + s) * 384 + h * DH + d;
        sq[s][d] = qkv[base];
        sk[s][d] = qkv[base + 128];
        sv[s][d] = qkv[base + 256];
    }
    __syncthreads();
    if (tid < NS) {
        float sc[NS];
        float mx = -1e30f;
        for (int kk = 0; kk < NS; ++kk) {
            float s = 0.f;
#pragma unroll
            for (int d = 0; d < DH; ++d) s += sq[tid][d] * sk[kk][d];
            s *= 0.25f;                       // 1/sqrt(16)
            sc[kk] = s;
            mx = fmaxf(mx, s);
        }
        float sum = 0.f;
        for (int kk = 0; kk < NS; ++kk) { sc[kk] = expf(sc[kk] - mx); sum += sc[kk]; }
        float inv = 1.0f / sum;
#pragma unroll
        for (int d = 0; d < DH; ++d) {
            float a = 0.f;
            for (int kk = 0; kk < NS; ++kk) a += sc[kk] * sv[kk][d];
            o16[((size_t)b * NS + tid) * EMB + h * DH + d] = (_Float16)(a * inv);
        }
    }
}

// ---------------- PEPS boundary contraction (chi=16 bond cap) -------------
__device__ __forceinline__ int udim(int i) { return i == 0 ? 1 : 4; }
__device__ __forceinline__ int ddim(int i) { return i == 5 ? 1 : 4; }
__device__ __forceinline__ int ldim(int j) { return j == 0 ? 1 : 4; }
__device__ __forceinline__ int rdim(int j) { return j == 5 ? 1 : 4; }

__device__ __forceinline__ float site_elem(const float* vb, const int* sx,
                                           int i, int j, int u, int d, int l, int r) {
    int k = i * 6 + j;
    int g = ((u * ddim(i) + d) * ldim(j) + l) * rdim(j) + r;
    return vb[c_offs[k] + 2 * g + sx[k]];
}

__global__ __launch_bounds__(128)
void k_peps(const int* __restrict__ x, const float* __restrict__ vecs,
            float* __restrict__ out) {
    __builtin_amdgcn_s_wait_tensorcnt(0);
    int b = blockIdx.x;
    const float* vb = vecs + (size_t)b * TOTAL;
    __shared__ int   sx[36];
    __shared__ float L[6][1024];     // lmps row tensors, fixed stride (16,16,4)
    __shared__ float R[6][1024];     // rmps row tensors
    __shared__ float Tmp[1024];
    __shared__ float Mz[256];
    __shared__ float Tz[1024];
    int tid = threadIdx.x;
    if (tid < 36) sx[tid] = x[b * 36 + tid];
    __syncthreads();

    int aL[6], cL[6], aR[6], cR[6];
    for (int i = 0; i < 6; ++i) {
        aL[i] = udim(i); cL[i] = ddim(i);
        aR[i] = udim(i); cR[i] = ddim(i);
    }

    // ---- left boundary: init col 0 ----
    for (int i = 0; i < 6; ++i) {
        int tot = aL[i] * cL[i] * 4;
        for (int t = tid; t < tot; t += blockDim.x) {
            int p = t & 3, c = (t >> 2) % cL[i], a = (t >> 2) / cL[i];
            L[i][(a * 16 + c) * 4 + p] = site_elem(vb, sx, i, 0, a, c, 0, p);
        }
    }
    // ---- right boundary: init col 5 ----
    for (int i = 0; i < 6; ++i) {
        int tot = aR[i] * cR[i] * 4;
        for (int t = tid; t < tot; t += blockDim.x) {
            int p = t & 3, c = (t >> 2) % cR[i], a = (t >> 2) / cR[i];
            R[i][(a * 16 + c) * 4 + p] = site_elem(vb, sx, i, 5, a, c, p, 0);
        }
    }
    __syncthreads();

    // ---- absorb columns 1,2 into left boundary ----
    for (int j = 1; j <= 2; ++j) {
        for (int i = 0; i < 6; ++i) {
            int ud = udim(i), dd = ddim(i);
            int na = min(16, aL[i] * ud), nc = min(16, cL[i] * dd);
            int tot = na * nc * 4;
            for (int t = tid; t < tot; t += blockDim.x) {
                int r = t & 3, C = (t >> 2) % nc, A = (t >> 2) / nc;
                int a = A / ud, u = A % ud, c = C / dd, d = C % dd;
                float s = 0.f;
#pragma unroll
                for (int l = 0; l < 4; ++l)
                    s += L[i][(a * 16 + c) * 4 + l] * site_elem(vb, sx, i, j, u, d, l, r);
                Tmp[t] = s;
            }
            __syncthreads();
            for (int t = tid; t < tot; t += blockDim.x) {
                int r = t & 3, C = (t >> 2) % nc, A = (t >> 2) / nc;
                L[i][(A * 16 + C) * 4 + r] = Tmp[t];
            }
            __syncthreads();
            aL[i] = na; cL[i] = nc;
        }
    }

    // ---- absorb columns 4,3 into right boundary ----
    for (int j = 4; j >= 3; --j) {
        for (int i = 0; i < 6; ++i) {
            int ud = udim(i), dd = ddim(i);
            int na = min(16, aR[i] * ud), nc = min(16, cR[i] * dd);
            int tot = na * nc * 4;
            for (int t = tid; t < tot; t += blockDim.x) {
                int lo = t & 3, C = (t >> 2) % nc, A = (t >> 2) / nc;
                int a = A / ud, u = A % ud, c = C / dd, d = C % dd;
                float s = 0.f;
#pragma unroll
                for (int r = 0; r < 4; ++r)
                    s += R[i][(a * 16 + c) * 4 + r] * site_elem(vb, sx, i, j, u, d, lo, r);
                Tmp[t] = s;
            }
            __syncthreads();
            for (int t = tid; t < tot; t += blockDim.x) {
                int lo = t & 3, C = (t >> 2) % nc, A = (t >> 2) / nc;
                R[i][(A * 16 + C) * 4 + lo] = Tmp[t];
            }
            __syncthreads();
            aR[i] = na; cR[i] = nc;
        }
    }

    // ---- zip: M <- einsum('xy,xap,ybp->ab', M, L[i], R[i]) ----
    if (tid == 0) Mz[0] = 1.0f;
    __syncthreads();
    for (int i = 0; i < 6; ++i) {
        // step1: T[y,a,p] = sum_x M[x,y] * L[i][x,a,p]
        int tot = aR[i] * cL[i] * 4;
        for (int t = tid; t < tot; t += blockDim.x) {
            int p = t & 3, a = (t >> 2) % cL[i], y = (t >> 2) / cL[i];
            float s = 0.f;
            for (int xx = 0; xx < aL[i]; ++xx)
                s += Mz[xx * 16 + y] * L[i][(xx * 16 + a) * 4 + p];
            Tz[(y * 16 + a) * 4 + p] = s;
        }
        __syncthreads();
        // step2: M[a,b] = sum_{y,p} T[y,a,p] * R[i][y,b,p]
        tot = cL[i] * cR[i];
        for (int t = tid; t < tot; t += blockDim.x) {
            int bb = t % cR[i], a = t / cR[i];
            float s = 0.f;
            for (int y = 0; y < aR[i]; ++y)
#pragma unroll
                for (int p = 0; p < 4; ++p)
                    s += Tz[(y * 16 + a) * 4 + p] * R[i][(y * 16 + bb) * 4 + p];
            Mz[a * 16 + bb] = s;
        }
        __syncthreads();
    }
    if (tid == 0) out[b] = Mz[0];
}

// ---------------- host: workspace layout + launches ----------------
extern "C" void kernel_launch(void* const* d_in, const int* in_sizes, int n_in,
                              void* d_out, int out_size, void* d_ws, size_t ws_size,
                              hipStream_t stream) {
    (void)in_sizes; (void)n_in; (void)out_size; (void)ws_size;
    const int*   x    = (const int*)d_in[0];
    const float* ftn  = (const float*)d_in[1];
    const float* emb  = (const float*)d_in[2];
    const float* pos  = (const float*)d_in[3];
    const float* Wqkv = (const float*)d_in[4];
    const float* bqkv = (const float*)d_in[5];
    const float* Wo   = (const float*)d_in[6];
    const float* bo   = (const float*)d_in[7];
    const float* W1   = (const float*)d_in[8];
    const float* b1   = (const float*)d_in[9];
    const float* W2   = (const float*)d_in[10];
    const float* b2   = (const float*)d_in[11];
    float* out = (float*)d_out;

    char* ws = (char*)d_ws;
    _Float16* WqkvT = (_Float16*)(ws + 0);                    // 384x128
    _Float16* WoT   = (_Float16*)(ws + 98304);                // 128x128
    _Float16* W1T   = (_Float16*)(ws + 131072);               // 512x128
    _Float16* W2T   = (_Float16*)(ws + 262144);               // 512x512
    float*    hf32  = (float*)   (ws + 786432);               // [M,128] f32
    _Float16* hf16  = (_Float16*)(ws + 38535168);             // [M,128] f16
    _Float16* o16   = (_Float16*)(ws + 57409536);             // [M,128] f16
    _Float16* h2f16 = (_Float16*)(ws + 76283904);             // [M,128] f16
    float*    qkv   = (float*)   (ws + 95158272);             // [M,384] f32
    _Float16* hid16 = (_Float16*)(ws + 95158272);             // alias (qkv dead)
    float*    vecs  = (float*)   (ws + 208404480);            // [B,TOTAL] f32

    // weight convert+transpose
    k_convT<<<(128 * 384 + 255) / 256, 256, 0, stream>>>(Wqkv, WqkvT, 128, 384);
    k_convT<<<(128 * 128 + 255) / 256, 256, 0, stream>>>(Wo, WoT, 128, 128);
    k_convT<<<(128 * 512 + 255) / 256, 256, 0, stream>>>(W1, W1T, 128, 512);
    k_convT<<<(512 * 512 + 255) / 256, 256, 0, stream>>>(W2, W2T, 512, 512);

    // embed + pos
    k_embed<<<(MTOK * EMB + 255) / 256, 256, 0, stream>>>(x, emb, pos, hf32, hf16);

    // qkv = h @ Wqkv + bqkv           (M=73728, N=384, K=128) : 4608*12 wave-tiles
    k_gemm<384, 128, 0><<<(4608 * 12) / 8, 256, 0, stream>>>(
        hf16, WqkvT, bqkv, nullptr, nullptr, qkv, nullptr);
    // attention
    k_attn<<<BATCH * HEADS, 64, 0, stream>>>(qkv, o16);
    // h2 = h + o @ Wo + bo            (M=73728, N=128, K=128) : 4608*4 wave-tiles
    k_gemm<128, 128, 1><<<(4608 * 4) / 8, 256, 0, stream>>>(
        o16, WoT, bo, hf32, nullptr, nullptr, h2f16);
    // hid = gelu(h2 @ W1 + b1)        (M=73728, N=512, K=128) : 4608*16 wave-tiles
    k_gemm<512, 128, 2><<<(4608 * 16) / 8, 256, 0, stream>>>(
        h2f16, W1T, b1, nullptr, nullptr, nullptr, hid16);
    // vecs = ftn + eta * trim(hid @ W2 + b2)   (M=73728, N=512, K=512)
    k_gemm<512, 512, 3><<<(4608 * 16) / 8, 256, 0, stream>>>(
        hid16, W2T, b2, nullptr, ftn, vecs, nullptr);
    // PEPS contraction: one workgroup per sample
    k_peps<<<BATCH, 128, 0, stream>>>(x, vecs, out);
}
